// Matcher_7404523618877
// MI455X (gfx1250) — compile-verified, matrix operand order checked by reading, not run on hardware
//
#include <hip/hip_runtime.h>

// ---------------------------------------------------------------------------
// Matcher (few-shot matching network) for MI455X / gfx1250.
// Heavy GEMMs use v_wmma_f32_16x16x32_f16 (f16 in, f32 accumulate).
// D=128, NB=30, STEPS=4, B=4096, FEW=5.
// ---------------------------------------------------------------------------

#define DD    128
#define NBN   30
#define NSTEP 4
#define BQ    4096
#define FEWN  5

typedef _Float16 half_t;
typedef __attribute__((ext_vector_type(16))) _Float16 v16h;
typedef __attribute__((ext_vector_type(8)))  _Float16 v8h;
typedef __attribute__((ext_vector_type(8)))  float    v8f;

// -------------------------- WMMA fragment helpers --------------------------
// 16-bit A/B operand packing (wave32): lane l, hi = l>>4 covers k = 8*hi + j
// (j=0..7) and k = 16 + 8*hi + j (j=0..7)  -> two contiguous 16B chunks.
// C/D f32: lane l -> N = (l&15), VGPR r -> M = r + 8*(l>>4).

__device__ __forceinline__ v8f v8f_zero() {
  v8f z;
#pragma unroll
  for (int i = 0; i < 8; ++i) z[i] = 0.0f;
  return z;
}

__device__ __forceinline__ v16h frag_f16(const half_t* rowptr, int kbase, int hi) {
  v8h lo = *(const v8h*)(rowptr + kbase + 8 * hi);
  v8h hh = *(const v8h*)(rowptr + kbase + 16 + 8 * hi);
  return __builtin_shufflevector(lo, hh, 0, 1, 2, 3, 4, 5, 6, 7,
                                         8, 9, 10, 11, 12, 13, 14, 15);
}

__device__ __forceinline__ v8f wmma16(v16h a, v16h b, v8f c) {
  // D = A(16x32 f16) * B(32x16 f16) + C(16x16 f32)
  return __builtin_amdgcn_wmma_f32_16x16x32_f16(false, a, false, b,
                                                (short)0, c, false, false);
}

__device__ __forceinline__ float sigmoid_(float x) {
  return 1.0f / (1.0f + expf(-x));
}

// 8-lane tree reduction (lane groups of 8; masks stay inside a wave32 half)
__device__ __forceinline__ float redux8(float v) {
  v += __shfl_xor(v, 1);
  v += __shfl_xor(v, 2);
  v += __shfl_xor(v, 4);
  return v;
}

// -------------------------- weight f32 -> f16 copy --------------------------
__global__ void cvt_f32_to_f16(const float* __restrict__ src,
                               half_t* __restrict__ dst, int n) {
  int i = blockIdx.x * blockDim.x + threadIdx.x;
  if (i < n) dst[i] = (half_t)src[i];
}

// ------------------------------ neighbor encoder ----------------------------
// One block per (row n). Computes mean & var attention paths, writes 128 cols
// at colOff of out_mean/out_var (row stride 256).
// nub/nvub are additive constants inside a softmax -> dropped (invariant).
__global__ __launch_bounds__(128) void neighbor_encoder_kernel(
    const int* __restrict__ conn,     // [n, NB, 2]
    const float* __restrict__ emb,    // [NSYM+1, 128]
    const float* __restrict__ embv,   // [NSYM+1, 128]
    const half_t* __restrict__ NWH,   // [128, 256] f16
    const float* __restrict__ nWb,    // [128]
    const float* __restrict__ nu,     // [128]
    const half_t* __restrict__ NVWH,  // [128, 256] f16
    const float* __restrict__ nvWb,   // [128]
    const float* __restrict__ nvu,    // [128]
    float* __restrict__ outM,         // [n, 256]
    float* __restrict__ outV,         // [n, 256]
    int colOff) {
  __shared__ half_t catM[32][256];    // [rel_e | ent_e] f16, rows 30,31 zero
  __shared__ half_t catV[32][256];
  __shared__ float  entE[NBN][128];   // ent_e f32 (pooled by BOTH paths)
  __shared__ float  aM[32], aV[32];
  __shared__ float  attM[NBN], attV[NBN];

  const int tid  = threadIdx.x;
  const int lane = tid & 31;
  const int wave = tid >> 5;
  const int hi   = (lane >> 4) & 1;
  const int l15  = lane & 15;
  const int n    = blockIdx.x;

  // zero pad rows 30,31
  for (int i = tid; i < 2 * 256; i += 128) {
    catM[30 + (i >> 8)][i & 255] = (half_t)0.0f;
    catV[30 + (i >> 8)][i & 255] = (half_t)0.0f;
  }

  // gather embeddings: thread tid covers feature dim d = tid
  const int* cr = conn + (size_t)n * (NBN * 2);
  for (int k = 0; k < NBN; ++k) {
    const size_t rel = (size_t)cr[2 * k];
    const size_t ent = (size_t)cr[2 * k + 1];
    float re = emb[rel * DD + tid];
    float ee = emb[ent * DD + tid];
    catM[k][tid]       = (half_t)re;
    catM[k][128 + tid] = (half_t)ee;
    entE[k][tid]       = ee;
    catV[k][tid]       = (half_t)embv[rel * DD + tid];
    catV[k][128 + tid] = (half_t)embv[ent * DD + tid];
  }
  __syncthreads();

  // tanh(cat @ W^T + b) @ u : wave w owns (path = w>>1, mtile = w&1),
  // loops the 8 N-tiles, accumulating per-row logits in registers.
  {
    const int path = wave >> 1;
    const int mt   = wave & 1;
    const half_t (*cat)[256] = path ? catV : catM;
    const half_t* W  = path ? NVWH : NWH;
    const float*  bb = path ? nvWb : nWb;
    const float*  uu = path ? nvu  : nu;
    float*        aa = path ? aV   : aM;

    const int mrow = mt * 16 + l15;      // A row
    float aPart[8];
#pragma unroll
    for (int r = 0; r < 8; ++r) aPart[r] = 0.0f;

#pragma unroll
    for (int nt = 0; nt < 8; ++nt) {
      const int ncol = nt * 16 + l15;    // output col (0..127)
      v8f acc = v8f_zero();
#pragma unroll
      for (int kt = 0; kt < 8; ++kt) {   // K = 256
        v16h af = frag_f16(&cat[mrow][0], kt * 32, hi);
        v16h bf = frag_f16(W + (size_t)ncol * 256, kt * 32, hi);
        acc = wmma16(af, bf, acc);
      }
      const float bias = bb[ncol];
      const float u    = uu[ncol];
#pragma unroll
      for (int r = 0; r < 8; ++r)
        aPart[r] += tanhf(acc[r] + bias) * u;
    }
    // reduce over N (16 lanes of each half hold distinct N, same M set)
#pragma unroll
    for (int r = 0; r < 8; ++r) {
      float v = aPart[r];
      v += __shfl_xor(v, 1);
      v += __shfl_xor(v, 2);
      v += __shfl_xor(v, 4);
      v += __shfl_xor(v, 8);
      if (l15 == 0) aa[mt * 16 + r + 8 * hi] = v;
    }
  }
  __syncthreads();

  // softmax over 30 neighbors (one thread per path)
  if (tid < 2) {
    float* aa  = tid ? aV : aM;
    float* att = tid ? attV : attM;
    float m = aa[0];
    for (int k = 1; k < NBN; ++k) m = fmaxf(m, aa[k]);
    float s = 0.0f;
    for (int k = 0; k < NBN; ++k) { float e = expf(aa[k] - m); att[k] = e; s += e; }
    float inv = 1.0f / s;
    for (int k = 0; k < NBN; ++k) att[k] *= inv;
  }
  __syncthreads();

  // attention pooling over ent_e (both paths pool the mean-table embeds)
  {
    float oM = 0.0f, oV = 0.0f;
#pragma unroll 5
    for (int k = 0; k < NBN; ++k) {
      float e = entE[k][tid];
      oM += attM[k] * e;
      oV += attV[k] * e;
    }
    outM[(size_t)n * 256 + colOff + tid] = tanhf(oM);
    outV[(size_t)n * 256 + colOff + tid] = tanhf(oV);
  }
}

// ------------------------------ support encoder -----------------------------
// LayerNorm(relu(x@p1W^T+p1b)@p2W^T+p2b + x), 16 rows per block.
__global__ __launch_bounds__(128) void support_encoder_kernel(
    const float* __restrict__ x,       // [nrows, 256]
    const half_t* __restrict__ P1H,    // [512, 256] f16
    const half_t* __restrict__ P2H,    // [256, 512] f16
    const float* __restrict__ p1b,     // [512]
    const float* __restrict__ p2b,     // [256]
    const float* __restrict__ lna,     // [256]
    const float* __restrict__ lnb,     // [256]
    float* __restrict__ out,           // [nrows, 256]
    int nrows) {
  __shared__ float  Xs[16][256];
  __shared__ half_t XsH[16][256];
  __shared__ half_t H1[16][512];
  __shared__ float  Ho[16][256];

  const int tid  = threadIdx.x;
  const int lane = tid & 31;
  const int wave = tid >> 5;
  const int hi   = (lane >> 4) & 1;
  const int l15  = lane & 15;
  const int rb   = blockIdx.x * 16;

  for (int i = tid; i < 16 * 256; i += 128) {
    int r = i >> 8, c = i & 255;
    float v = (rb + r < nrows) ? x[(size_t)(rb + r) * 256 + c] : 0.0f;
    Xs[r][c]  = v;
    XsH[r][c] = (half_t)v;
  }
  __syncthreads();

  // GEMM1: [16,256] x [256,512] -> relu -> H1 (f16). 8 N-tiles per wave.
#pragma unroll
  for (int t = 0; t < 8; ++t) {
    const int ncol = (wave * 8 + t) * 16 + l15;
    v8f acc = v8f_zero();
#pragma unroll
    for (int kt = 0; kt < 8; ++kt) {
      v16h af = frag_f16(&XsH[l15][0], kt * 32, hi);
      v16h bf = frag_f16(P1H + (size_t)ncol * 256, kt * 32, hi);
      acc = wmma16(af, bf, acc);
    }
    const float bias = p1b[ncol];
#pragma unroll
    for (int r = 0; r < 8; ++r) {
      float v = fmaxf(acc[r] + bias, 0.0f);
      H1[r + 8 * hi][ncol] = (half_t)v;
    }
  }
  __syncthreads();

  // GEMM2: [16,512] x [512,256] + bias + residual. 4 N-tiles per wave.
#pragma unroll
  for (int t = 0; t < 4; ++t) {
    const int ncol = (wave * 4 + t) * 16 + l15;
    v8f acc = v8f_zero();
#pragma unroll
    for (int kt = 0; kt < 16; ++kt) {
      v16h af = frag_f16(&H1[l15][0], kt * 32, hi);
      v16h bf = frag_f16(P2H + (size_t)ncol * 512, kt * 32, hi);
      acc = wmma16(af, bf, acc);
    }
    const float bias = p2b[ncol];
#pragma unroll
    for (int r = 0; r < 8; ++r) {
      int M = r + 8 * hi;
      Ho[M][ncol] = acc[r] + bias + Xs[M][ncol];
    }
  }
  __syncthreads();

  // LayerNorm per row (Bessel-corrected std, eps=1e-3): 8 lanes per row,
  // shfl-tree reductions, all 128 lanes active.
  {
    const int r  = tid >> 3;
    const int t8 = tid & 7;
    const int c0 = t8 * 32;
    float part = 0.0f;
    for (int c = c0; c < c0 + 32; ++c) part += Ho[r][c];
    const float mu = redux8(part) * (1.0f / 256.0f);
    float vp = 0.0f;
    for (int c = c0; c < c0 + 32; ++c) { float d = Ho[r][c] - mu; vp += d * d; }
    const float inv = 1.0f / (sqrtf(redux8(vp) / 255.0f) + 0.001f);
    const int grow = rb + r;
    if (grow < nrows) {
      for (int c = c0; c < c0 + 32; ++c)
        out[(size_t)grow * 256 + c] = (Ho[r][c] - mu) * inv * lna[c] + lnb[c];
    }
  }
}

// ------------------------------- query encoder ------------------------------
// Fully fused 4-step LSTM + support attention + final score for 16 rows/block.
// Wave w owns hidden columns [w*128, w*128+128); c-state lives in registers.
__global__ __launch_bounds__(128) void query_encoder_kernel(
    const float* __restrict__ qg,     // [nrows, 256]   (support-encoded query)
    const float* __restrict__ sgc,    // [5, 256]       (support-encoded support)
    const half_t* __restrict__ WihH,  // [2048, 256] f16
    const half_t* __restrict__ WhhH,  // [2048, 512] f16
    const float* __restrict__ bih,    // [2048]
    const float* __restrict__ bhh,    // [2048]
    float* __restrict__ out,          // [nrows]
    int nrows) {
  __shared__ float  sg[FEWN][256];
  __shared__ float  smean[256];
  __shared__ float  Xq[16][256];      // query rows f32 (residual add)
  __shared__ half_t XqH[16][256];     // query rows f16 (GEMM A operand)
  __shared__ half_t hr[16][512];      // [h | r] f16 (GEMM A operand)
  __shared__ float  hbuf[16][256];    // h (f32, for attention + score)
  __shared__ float  attl[16][FEWN];

  const int tid  = threadIdx.x;
  const int lane = tid & 31;
  const int wave = tid >> 5;
  const int hi   = (lane >> 4) & 1;
  const int l15  = lane & 15;
  const int rb   = blockIdx.x * 16;
  const int hc   = wave * 128;        // this wave's hidden column base

  for (int i = tid; i < FEWN * 256; i += 128) sg[i >> 8][i & 255] = sgc[i];
  for (int i = tid; i < 16 * 256; i += 128) {
    int r = i >> 8, c = i & 255;
    float v = (rb + r < nrows) ? qg[(size_t)(rb + r) * 256 + c] : 0.0f;
    Xq[r][c]  = v;
    XqH[r][c] = (half_t)v;
  }
  for (int i = tid; i < 16 * 512; i += 128) (&hr[0][0])[i] = (half_t)0.0f;
  __syncthreads();
  for (int c = tid; c < 256; c += 128) {
    float s = 0.0f;
    for (int k = 0; k < FEWN; ++k) s += sg[k][c];
    smean[c] = s * 0.2f;
  }

  float creg[8][8];                   // LSTM cell state, this wave's 128 cols
#pragma unroll
  for (int a = 0; a < 8; ++a)
#pragma unroll
    for (int b = 0; b < 8; ++b) creg[a][b] = 0.0f;

  for (int step = 0; step < NSTEP; ++step) {
    __syncthreads();                  // hr / smean ready

    // gates = Xq @ Wih^T + hr @ Whh^T + bih + bhh (torch order: i,f,g,o)
#pragma unroll
    for (int nt = 0; nt < 8; ++nt) {
      const int ncol = hc + nt * 16 + l15;          // hidden col (0..511)
      __builtin_prefetch(WihH + (size_t)ncol * 256, 0, 1);
      v8f acc[4];
#pragma unroll
      for (int g = 0; g < 4; ++g) acc[g] = v8f_zero();

#pragma unroll
      for (int kt = 0; kt < 8; ++kt) {              // K = 256 (input)
        v16h af = frag_f16(&XqH[l15][0], kt * 32, hi);
#pragma unroll
        for (int g = 0; g < 4; ++g) {
          v16h bf = frag_f16(WihH + (size_t)(g * 512 + ncol) * 256, kt * 32, hi);
          acc[g] = wmma16(af, bf, acc[g]);
        }
      }
#pragma unroll
      for (int kt = 0; kt < 16; ++kt) {             // K = 512 (hidden)
        v16h af = frag_f16(&hr[l15][0], kt * 32, hi);
#pragma unroll
        for (int g = 0; g < 4; ++g) {
          v16h bf = frag_f16(WhhH + (size_t)(g * 512 + ncol) * 512, kt * 32, hi);
          acc[g] = wmma16(af, bf, acc[g]);
        }
      }
      const float bI = bih[0 * 512 + ncol] + bhh[0 * 512 + ncol];
      const float bF = bih[1 * 512 + ncol] + bhh[1 * 512 + ncol];
      const float bG = bih[2 * 512 + ncol] + bhh[2 * 512 + ncol];
      const float bO = bih[3 * 512 + ncol] + bhh[3 * 512 + ncol];
#pragma unroll
      for (int r = 0; r < 8; ++r) {
        float iv = sigmoid_(acc[0][r] + bI);
        float fv = sigmoid_(acc[1][r] + bF);
        float gv = tanhf(acc[2][r] + bG);
        float ov = sigmoid_(acc[3][r] + bO);
        float cc = fv * creg[nt][r] + iv * gv;
        creg[nt][r] = cc;
        float hro = ov * tanhf(cc);
        if (hc < 256) {               // waves 0,1: h = query + h_r_[:, :256]
          int M = r + 8 * hi;
          hbuf[M][ncol] = Xq[M][ncol] + hro;
        }
      }
    }
    __syncthreads();                  // hbuf ready; everyone done reading hr

    // attention logits: h @ s_g^T, 8 lanes per row with shfl-tree reduction
    {
      const int r  = tid >> 3;
      const int t8 = tid & 7;
      const int c0 = t8 * 32;
#pragma unroll
      for (int s = 0; s < FEWN; ++s) {
        float p = 0.0f;
        for (int c = c0; c < c0 + 32; ++c) p += hbuf[r][c] * sg[s][c];
        p = redux8(p);
        if (t8 == 0) attl[r][s] = p;
      }
    }
    __syncthreads();
    if (tid < 16) {
      float m = attl[tid][0];
      for (int s = 1; s < FEWN; ++s) m = fmaxf(m, attl[tid][s]);
      float sum = 0.0f;
      float e[FEWN];
      for (int s = 0; s < FEWN; ++s) { e[s] = expf(attl[tid][s] - m); sum += e[s]; }
      float inv = 1.0f / sum;
      for (int s = 0; s < FEWN; ++s) attl[tid][s] = e[s] * inv;
    }
    __syncthreads();

    // hr = [h, attn @ s_g]  (f16 for next step's GEMM A)
    for (int i = tid; i < 16 * 256; i += 128) {
      int r = i >> 8, c = i & 255;
      hr[r][c] = (half_t)hbuf[r][c];
      float rv = 0.0f;
#pragma unroll
      for (int s = 0; s < FEWN; ++s) rv += attl[r][s] * sg[s][c];
      hr[r][256 + c] = (half_t)rv;
    }
  }
  __syncthreads();

  // score = h @ s_mean, 8 lanes per row
  {
    const int r  = tid >> 3;
    const int t8 = tid & 7;
    const int c0 = t8 * 32;
    float p = 0.0f;
    for (int c = c0; c < c0 + 32; ++c) p += hbuf[r][c] * smean[c];
    p = redux8(p);
    if (t8 == 0 && rb + r < nrows) out[rb + r] = p;
  }
}

// --------------------------------- launcher ---------------------------------
extern "C" void kernel_launch(void* const* d_in, const int* in_sizes, int n_in,
                              void* d_out, int out_size, void* d_ws,
                              size_t ws_size, hipStream_t stream) {
  (void)in_sizes; (void)n_in; (void)out_size; (void)ws_size;

  const float* emb    = (const float*)d_in[0];
  const float* embv   = (const float*)d_in[1];
  const float* nW     = (const float*)d_in[2];
  const float* nWb    = (const float*)d_in[3];
  const float* nu     = (const float*)d_in[4];
  // d_in[5] = nub: additive constant inside softmax -> invariant, unused
  const float* nvW    = (const float*)d_in[6];
  const float* nvWb   = (const float*)d_in[7];
  const float* nvu    = (const float*)d_in[8];
  // d_in[9] = nvub: unused (softmax invariant)
  const float* p1W    = (const float*)d_in[10];
  const float* p1b    = (const float*)d_in[11];
  const float* p2W    = (const float*)d_in[12];
  const float* p2b    = (const float*)d_in[13];
  const float* lna    = (const float*)d_in[14];
  const float* lnb    = (const float*)d_in[15];
  const float* Wih    = (const float*)d_in[16];
  const float* Whh    = (const float*)d_in[17];
  const float* bihp   = (const float*)d_in[18];
  const float* bhhp   = (const float*)d_in[19];
  const int*   q_left  = (const int*)d_in[20];
  const int*   q_right = (const int*)d_in[21];
  const int*   s_left  = (const int*)d_in[22];
  const int*   s_right = (const int*)d_in[23];
  float* out = (float*)d_out;

  char*  ws  = (char*)d_ws;
  size_t off = 0;
  auto take = [&](size_t bytes) -> void* {
    void* p = (void*)(ws + off);
    off += (bytes + 255) & ~(size_t)255;
    return p;
  };

  half_t* WihH = (half_t*)take(2048 * 256 * sizeof(half_t));
  half_t* WhhH = (half_t*)take(2048 * 512 * sizeof(half_t));
  half_t* P1H  = (half_t*)take(512 * 256 * sizeof(half_t));
  half_t* P2H  = (half_t*)take(256 * 512 * sizeof(half_t));
  half_t* NWH  = (half_t*)take(128 * 256 * sizeof(half_t));
  half_t* NVWH = (half_t*)take(128 * 256 * sizeof(half_t));
  float* QUERY  = (float*)take((size_t)BQ * 256 * sizeof(float));
  float* QUERYV = (float*)take((size_t)BQ * 256 * sizeof(float));
  float* SUPP   = (float*)take(FEWN * 256 * sizeof(float));
  float* SUPPV  = (float*)take(FEWN * 256 * sizeof(float));
  float* QG     = (float*)take((size_t)BQ * 256 * sizeof(float));
  float* SG     = (float*)take(FEWN * 256 * sizeof(float));

  // 1) weight conversions (f32 -> f16, kept in L2-resident workspace)
  auto cvt = [&](const float* s, half_t* d, int n) {
    cvt_f32_to_f16<<<(n + 255) / 256, 256, 0, stream>>>(s, d, n);
  };
  cvt(Wih, WihH, 2048 * 256);
  cvt(Whh, WhhH, 2048 * 512);
  cvt(p1W, P1H, 512 * 256);
  cvt(p2W, P2H, 256 * 512);
  cvt(nW,  NWH,  128 * 256);
  cvt(nvW, NVWH, 128 * 256);

  // 2) neighbor encoder: query (left/right halves) and support
  neighbor_encoder_kernel<<<BQ, 128, 0, stream>>>(
      q_left, emb, embv, NWH, nWb, nu, NVWH, nvWb, nvu, QUERY, QUERYV, 0);
  neighbor_encoder_kernel<<<BQ, 128, 0, stream>>>(
      q_right, emb, embv, NWH, nWb, nu, NVWH, nvWb, nvu, QUERY, QUERYV, 128);
  neighbor_encoder_kernel<<<FEWN, 128, 0, stream>>>(
      s_left, emb, embv, NWH, nWb, nu, NVWH, nvWb, nvu, SUPP, SUPPV, 0);
  neighbor_encoder_kernel<<<FEWN, 128, 0, stream>>>(
      s_right, emb, embv, NWH, nWb, nu, NVWH, nvWb, nvu, SUPP, SUPPV, 128);

  // 3) mean-path match
  support_encoder_kernel<<<1, 128, 0, stream>>>(
      SUPP, P1H, P2H, p1b, p2b, lna, lnb, SG, FEWN);
  support_encoder_kernel<<<BQ / 16, 128, 0, stream>>>(
      QUERY, P1H, P2H, p1b, p2b, lna, lnb, QG, BQ);
  query_encoder_kernel<<<BQ / 16, 128, 0, stream>>>(
      QG, SG, WihH, WhhH, bihp, bhhp, out, BQ);

  // 4) var-path match (reuses SG/QG scratch; stream-ordered)
  support_encoder_kernel<<<1, 128, 0, stream>>>(
      SUPPV, P1H, P2H, p1b, p2b, lna, lnb, SG, FEWN);
  support_encoder_kernel<<<BQ / 16, 128, 0, stream>>>(
      QUERYV, P1H, P2H, p1b, p2b, lna, lnb, QG, BQ);
  query_encoder_kernel<<<BQ / 16, 128, 0, stream>>>(
      QG, SG, WihH, WhhH, bihp, bhhp, out + BQ, BQ);
}
